// Attention_52132313039121
// MI455X (gfx1250) — matrix operand order repaired and checked by reading, not compiled, not run on hardware
//
#include <hip/hip_runtime.h>
#include <hip/hip_bf16.h>

typedef _Float16 v16h __attribute__((ext_vector_type(16)));
typedef _Float16 v8h  __attribute__((ext_vector_type(8)));
typedef _Float16 v4h  __attribute__((ext_vector_type(4)));
typedef float    v8f  __attribute__((ext_vector_type(8)));
typedef float    v4f  __attribute__((ext_vector_type(4)));
typedef unsigned int u32x4 __attribute__((ext_vector_type(4)));
typedef int          i32x8 __attribute__((ext_vector_type(8)));
typedef int          i32x4 __attribute__((ext_vector_type(4)));

union V16 { v16h v; v8h h[2]; };

#define SEQ      2048
#define DIM      4096
#define NH       32
#define NKV      8
#define HD       128
#define KVDIM    1024
#define SCALE    0.08838834764831845f
#define LOG2_10000 13.287712379549449f

__device__ inline unsigned short hbits(_Float16 h) {
  union { _Float16 f; unsigned short u; } c; c.f = h; return c.u;
}

// ---------------------------------------------------------------------------
// f32 -> f16 bulk convert (4 elements/thread)
// ---------------------------------------------------------------------------
__global__ void cvt_f16_kernel(const float* __restrict__ src, _Float16* __restrict__ dst) {
  int i = (blockIdx.x * blockDim.x + threadIdx.x) * 4;
  v4f f = *(const v4f*)(src + i);
  v4h o;
  #pragma unroll
  for (int j = 0; j < 4; ++j) o[j] = (_Float16)f[j];
  *(v4h*)(dst + i) = o;
}

// ---------------------------------------------------------------------------
// Tiled WMMA GEMM: C[M,N](f32) = A[M,K](f16) @ B[K,N](f16), row-major.
// N,K compile-time -> immediate-offset stores. Double-buffered LDS, one
// barrier per 32-K step. Block tile 64x128; 8 waves in 2x4 of 32x32 tiles.
// ---------------------------------------------------------------------------
template<int N, int K>
__global__ __launch_bounds__(256) void gemm_f16_wmma(
    const _Float16* __restrict__ A, const _Float16* __restrict__ B,
    float* __restrict__ C) {
  __shared__ _Float16 As[2][64 * 32];    // [m][k]
  __shared__ _Float16 Bs[2][128 * 32];   // [n][k] (transposed staging)
  const int tid  = threadIdx.x;
  const int lane = tid & 31;
  const int wid  = tid >> 5;
  const int wm   = wid & 1;
  const int wn   = wid >> 1;
  const int lh   = lane & 15;
  const int lhh  = lane >> 4;
  const int bm   = blockIdx.y;
  const int bn   = blockIdx.x;

  // staging indices (precomputed)
  const int a_r = (tid * 8) >> 5, a_c = (tid * 8) & 31;            // 64x32, v8h each
  const int b_kr = (tid >> 4) * 2, b_n0 = (tid & 15) * 8;          // 2 k-rows x 8 n

  auto stage = [&](int buf, int kb) {
    v8h ga = *(const v8h*)(A + (size_t)(bm * 64 + a_r) * K + kb + a_c);
    *(v8h*)&As[buf][a_r * 32 + a_c] = ga;
    v8h g0 = *(const v8h*)(B + (size_t)(kb + b_kr) * N + bn * 128 + b_n0);
    v8h g1 = *(const v8h*)(B + (size_t)(kb + b_kr + 1) * N + bn * 128 + b_n0);
    #pragma unroll
    for (int i = 0; i < 8; ++i) {
      unsigned pk = (unsigned)hbits(g0[i]) | ((unsigned)hbits(g1[i]) << 16);
      *(unsigned*)&Bs[buf][(b_n0 + i) * 32 + b_kr] = pk;   // pack (k, k+1)
    }
  };

  v8f acc[2][2];
  #pragma unroll
  for (int i = 0; i < 2; ++i)
    #pragma unroll
    for (int j = 0; j < 2; ++j)
      #pragma unroll
      for (int r = 0; r < 8; ++r) acc[i][j][r] = 0.0f;

  stage(0, 0);
  __syncthreads();

  const int koff = lhh * 16;
  for (int it = 0; it < K / 32; ++it) {
    int cur = it & 1;
    if (it + 1 < K / 32) stage(cur ^ 1, (it + 1) * 32);

    V16 a[2], b[2];
    #pragma unroll
    for (int im = 0; im < 2; ++im) {
      int mrow = wm * 32 + im * 16 + lh;
      a[im].h[0] = *(const v8h*)&As[cur][mrow * 32 + koff];
      a[im].h[1] = *(const v8h*)&As[cur][mrow * 32 + koff + 8];
    }
    #pragma unroll
    for (int in2 = 0; in2 < 2; ++in2) {
      int nrow = wn * 32 + in2 * 16 + lh;
      b[in2].v = *(const v16h*)&Bs[cur][nrow * 32 + lhh * 16];
    }
    #pragma unroll
    for (int im = 0; im < 2; ++im)
      #pragma unroll
      for (int in2 = 0; in2 < 2; ++in2)
        acc[im][in2] = __builtin_amdgcn_wmma_f32_16x16x32_f16(
            false, a[im].v, false, b[in2].v, (short)0, acc[im][in2], false, false);
    __syncthreads();
  }

  #pragma unroll
  for (int im = 0; im < 2; ++im)
    #pragma unroll
    for (int in2 = 0; in2 < 2; ++in2) {
      float* cp = C + (size_t)(bm * 64 + wm * 32 + im * 16 + 8 * lhh) * N
                    + bn * 128 + wn * 32 + in2 * 16 + lh;
      #pragma unroll
      for (int r = 0; r < 8; ++r) cp[(size_t)r * N] = acc[im][in2][r];
    }
}

// ---------------------------------------------------------------------------
// RoPE Q: read f32 GEMM output, write f16 (only consumer is flash attn).
// ---------------------------------------------------------------------------
__global__ void rope_q_kernel(const float* __restrict__ q, _Float16* __restrict__ q16) {
  int idx = blockIdx.x * blockDim.x + threadIdx.x;   // SEQ*DIM/2 pairs
  int t  = idx >> 11;
  int j0 = (idx & 2047) * 2;
  const float* row = q + (size_t)t * DIM;
  float x0 = row[j0], x1 = row[j0 + 1];
  int i0 = j0 & 63;
  float inv0 = exp2f(-(float)i0       * (LOG2_10000 / 64.0f));
  float inv1 = exp2f(-(float)(i0 + 1) * (LOG2_10000 / 64.0f));
  float th0 = (float)t * inv0, th1 = (float)t * inv1;
  float o0 = x0 * cosf(th0) - x1 * sinf(th0);
  float o1 = x1 * cosf(th1) + x0 * sinf(th1);
  q16[(size_t)t * DIM + j0]     = (_Float16)o0;
  q16[(size_t)t * DIM + j0 + 1] = (_Float16)o1;
}

// RoPE K: f16 copy for attention + GQA-expanded f32 reference output.
__global__ void rope_k_kernel(const float* __restrict__ k, _Float16* __restrict__ k16,
                              float* __restrict__ kout) {
  int idx = blockIdx.x * blockDim.x + threadIdx.x;   // SEQ*KVDIM/2 pairs
  int t  = idx >> 9;
  int j0 = (idx & 511) * 2;
  int d0 = j0 & 127, kvh = j0 >> 7;
  const float* row = k + (size_t)t * KVDIM;
  float x0 = row[j0], x1 = row[j0 + 1];
  int i0 = j0 & 63;
  float inv0 = exp2f(-(float)i0       * (LOG2_10000 / 64.0f));
  float inv1 = exp2f(-(float)(i0 + 1) * (LOG2_10000 / 64.0f));
  float th0 = (float)t * inv0, th1 = (float)t * inv1;
  float o0 = x0 * cosf(th0) - x1 * sinf(th0);
  float o1 = x1 * cosf(th1) + x0 * sinf(th1);
  k16[(size_t)t * KVDIM + j0]     = (_Float16)o0;
  k16[(size_t)t * KVDIM + j0 + 1] = (_Float16)o1;
  #pragma unroll
  for (int rep = 0; rep < 4; ++rep) {
    float* dst = kout + ((size_t)(kvh * 4 + rep) * SEQ + t) * HD + d0;
    dst[0] = o0; dst[1] = o1;
  }
}

__global__ void v_expand_kernel(const float* __restrict__ v, _Float16* __restrict__ v16,
                                float* __restrict__ vout) {
  int idx = blockIdx.x * blockDim.x + threadIdx.x;   // SEQ*KVDIM
  int t = idx >> 10, c = idx & 1023;
  int kvh = c >> 7, d = c & 127;
  float val = v[(size_t)t * KVDIM + c];
  v16[(size_t)t * KVDIM + c] = (_Float16)val;
  #pragma unroll
  for (int rep = 0; rep < 4; ++rep)
    vout[((size_t)(kvh * 4 + rep) * SEQ + t) * HD + d] = val;
}

// ---------------------------------------------------------------------------
// TDM: 2D f16 tile load global->LDS via tensor descriptor (D#).
// 6-arg builtin form (clang-23 / amdgpu-toolchain): extra i32x8 before cpol.
// ---------------------------------------------------------------------------
#if defined(__has_builtin) && __has_builtin(__builtin_amdgcn_tensor_load_to_lds)
#define HAVE_TDM 1
__device__ inline void tdm_load_tile_f16(unsigned lds_off, const _Float16* gptr,
                                         int tile_w, int tile_h, int row_stride_elems) {
  unsigned long long ga = (unsigned long long)(uintptr_t)gptr;
  u32x4 g0;
  g0[0] = 1u;                                            // count=1 (valid, user)
  g0[1] = lds_off;                                       // lds_addr (bytes)
  g0[2] = (unsigned)ga;                                  // global_addr[31:0]
  g0[3] = (unsigned)((ga >> 32) & 0x1ffffffu) | (2u << 30);  // ga[56:32] | type=2
  i32x8 g1;
  g1[0] = 1 << 16;                                       // data_size=1 -> 2 bytes
  g1[1] = (tile_w & 0xffff) << 16;                       // tensor_dim0 lo16
  g1[2] = ((unsigned)tile_w >> 16) | ((tile_h & 0xffff) << 16); // dim0 hi | dim1 lo
  g1[3] = ((unsigned)tile_h >> 16) | ((tile_w & 0xffff) << 16); // dim1 hi | tile_dim0
  g1[4] = tile_h & 0xffff;                               // tile_dim1 (tile_dim2=0)
  g1[5] = row_stride_elems;                              // tensor_dim0_stride lo32
  g1[6] = 0;                                             // stride hi | dim1_stride lo
  g1[7] = 0;
  i32x4 gz4 = {0, 0, 0, 0};
  i32x8 gz8 = {0, 0, 0, 0, 0, 0, 0, 0};
  __builtin_amdgcn_tensor_load_to_lds(g0, g1, gz4, gz4, gz8, 0);
}
#else
#define HAVE_TDM 0
#endif

// ---------------------------------------------------------------------------
// Flash attention (no mask): grid (qtile=16, head=32), 8 waves/block.
// K tile staged by TDM (wave 0) while all threads stage V transposed.
// ---------------------------------------------------------------------------
__global__ __launch_bounds__(256) void flash_kernel(
    const _Float16* __restrict__ q16, const _Float16* __restrict__ k16,
    const _Float16* __restrict__ v16, _Float16* __restrict__ oh) {
  __shared__ _Float16 Ks[64 * 128];        // [key][d]
  __shared__ _Float16 Vt[128 * 64];        // [d][key]
  __shared__ _Float16 Ps[8 * 16 * 64];     // per-wave P staging
  const int tid = threadIdx.x, lane = tid & 31, wid = tid >> 5;
  const int lh = lane & 15, lhh = lane >> 4;
  const int head  = blockIdx.y;
  const int kvh   = head >> 2;
  const int qtile = blockIdx.x;
  const int koff  = lhh * 16;

  // Q fragments: 16 rows x 128 d -> 4 A-frags
  V16 qa[4];
  {
    int trow = qtile * 128 + wid * 16 + lh;
    const _Float16* qrow = q16 + (size_t)trow * DIM + head * HD;
    #pragma unroll
    for (int ks = 0; ks < 4; ++ks) {
      qa[ks].h[0] = *(const v8h*)(qrow + ks * 32 + koff);
      qa[ks].h[1] = *(const v8h*)(qrow + ks * 32 + koff + 8);
    }
  }

  float mrow[8], lrow[8];
  v8f accO[8];
  #pragma unroll
  for (int r = 0; r < 8; ++r) { mrow[r] = -1e30f; lrow[r] = 0.0f; }
  #pragma unroll
  for (int dt = 0; dt < 8; ++dt)
    #pragma unroll
    for (int r = 0; r < 8; ++r) accO[dt][r] = 0.0f;

  // V staging indices: 4 key-rows x 8 d per thread, b64-packed transpose
  const int v_r0 = (tid >> 4) * 4, v_c0 = (tid & 15) * 8;

  for (int kb = 0; kb < SEQ / 64; ++kb) {
    const _Float16* kblk = k16 + (size_t)(kb * 64) * KVDIM + kvh * HD;
    const _Float16* vblk = v16 + (size_t)(kb * 64) * KVDIM + kvh * HD;

#if HAVE_TDM
    if (wid == 0)
      tdm_load_tile_f16((unsigned)(uintptr_t)&Ks[0], kblk, 128, 64, KVDIM);
#else
    {
      int base = tid * 32, r = base >> 7, c = base & 127;
      #pragma unroll
      for (int j = 0; j < 4; ++j)
        *(v8h*)&Ks[r * 128 + c + j * 8] =
            *(const v8h*)(kblk + (size_t)r * KVDIM + c + j * 8);
    }
#endif
    // V transpose staging (overlaps TDM)
    {
      v8h rv[4];
      #pragma unroll
      for (int j = 0; j < 4; ++j)
        rv[j] = *(const v8h*)(vblk + (size_t)(v_r0 + j) * KVDIM + v_c0);
      #pragma unroll
      for (int i = 0; i < 8; ++i) {
        v4h pk;
        #pragma unroll
        for (int j = 0; j < 4; ++j) pk[j] = rv[j][i];
        *(v4h*)&Vt[(v_c0 + i) * 64 + v_r0] = pk;
      }
    }
    if (kb + 1 < SEQ / 64) {        // prefetch next block
      __builtin_prefetch(kblk + (size_t)64 * KVDIM + (tid >> 1) * KVDIM, 0, 0);
      __builtin_prefetch(vblk + (size_t)64 * KVDIM + (tid >> 1) * KVDIM, 0, 0);
    }
#if HAVE_TDM
    if (wid == 0) __builtin_amdgcn_s_wait_tensorcnt(0);
#endif
    __syncthreads();

    // S = Q @ K^T (16x64 per wave)
    v8f accS[4];
    #pragma unroll
    for (int nt = 0; nt < 4; ++nt)
      #pragma unroll
      for (int r = 0; r < 8; ++r) accS[nt][r] = 0.0f;
    #pragma unroll
    for (int ks = 0; ks < 4; ++ks)
      #pragma unroll
      for (int nt = 0; nt < 4; ++nt) {
        V16 bf;
        bf.v = *(const v16h*)&Ks[(nt * 16 + lh) * 128 + ks * 32 + lhh * 16];
        accS[nt] = __builtin_amdgcn_wmma_f32_16x16x32_f16(
            false, qa[ks].v, false, bf.v, (short)0, accS[nt], false, false);
      }
    #pragma unroll
    for (int nt = 0; nt < 4; ++nt)
      #pragma unroll
      for (int r = 0; r < 8; ++r) accS[nt][r] *= SCALE;

    // online softmax (rows live in 16-lane groups)
    float newm[8];
    #pragma unroll
    for (int r = 0; r < 8; ++r) {
      float v = fmaxf(fmaxf(accS[0][r], accS[1][r]), fmaxf(accS[2][r], accS[3][r]));
      #pragma unroll
      for (int sh = 1; sh < 16; sh <<= 1) v = fmaxf(v, __shfl_xor(v, sh, 16));
      newm[r] = fmaxf(mrow[r], v);
    }
    float psum[8];
    #pragma unroll
    for (int r = 0; r < 8; ++r) psum[r] = 0.0f;
    #pragma unroll
    for (int nt = 0; nt < 4; ++nt)
      #pragma unroll
      for (int r = 0; r < 8; ++r) {
        float pv = __expf(accS[nt][r] - newm[r]);
        accS[nt][r] = pv;
        psum[r] += pv;
      }
    #pragma unroll
    for (int r = 0; r < 8; ++r) {
      #pragma unroll
      for (int sh = 1; sh < 16; sh <<= 1) psum[r] += __shfl_xor(psum[r], sh, 16);
      float alpha = __expf(mrow[r] - newm[r]);
      lrow[r] = lrow[r] * alpha + psum[r];
      mrow[r] = newm[r];
      #pragma unroll
      for (int dt = 0; dt < 8; ++dt) accO[dt][r] *= alpha;
    }

    // P -> LDS f16 (per-wave region) -> A-frags
    _Float16* pw = &Ps[wid * 16 * 64];
    #pragma unroll
    for (int nt = 0; nt < 4; ++nt)
      #pragma unroll
      for (int r = 0; r < 8; ++r)
        pw[(r + 8 * lhh) * 64 + nt * 16 + lh] = (_Float16)accS[nt][r];
    V16 pa[2];
    #pragma unroll
    for (int ks2 = 0; ks2 < 2; ++ks2) {
      pa[ks2].h[0] = *(const v8h*)&pw[lh * 64 + ks2 * 32 + koff];
      pa[ks2].h[1] = *(const v8h*)&pw[lh * 64 + ks2 * 32 + koff + 8];
    }
    // O += P @ V
    #pragma unroll
    for (int ks2 = 0; ks2 < 2; ++ks2)
      #pragma unroll
      for (int dt = 0; dt < 8; ++dt) {
        V16 bf;
        bf.v = *(const v16h*)&Vt[(dt * 16 + lh) * 64 + ks2 * 32 + lhh * 16];
        accO[dt] = __builtin_amdgcn_wmma_f32_16x16x32_f16(
            false, pa[ks2].v, false, bf.v, (short)0, accO[dt], false, false);
      }
    __syncthreads();
  }

  // epilogue: O /= l, write f16 (feeds wo GEMM)
  #pragma unroll
  for (int r = 0; r < 8; ++r) {
    float inv = 1.0f / lrow[r];
    int t = qtile * 128 + wid * 16 + r + 8 * lhh;
    _Float16* dst = oh + (size_t)t * DIM + head * HD;
    #pragma unroll
    for (int dt = 0; dt < 8; ++dt) dst[dt * 16 + lh] = (_Float16)(accO[dt][r] * inv);
  }
}

// ---------------------------------------------------------------------------
extern "C" void kernel_launch(void* const* d_in, const int* in_sizes, int n_in,
                              void* d_out, int out_size, void* d_ws, size_t ws_size,
                              hipStream_t stream) {
  const float* x  = (const float*)d_in[0];
  const float* wq = (const float*)d_in[1];
  const float* wk = (const float*)d_in[2];
  const float* wv = (const float*)d_in[3];
  const float* wo = (const float*)d_in[4];

  float* out0 = (float*)d_out;
  float* kout = out0 + (size_t)SEQ * DIM;
  float* vout = kout + (size_t)NH * SEQ * HD;

  _Float16* xh  = (_Float16*)d_ws;
  _Float16* wqh = xh + (size_t)SEQ * DIM;            // reused for wo later
  _Float16* wkh = wqh + (size_t)DIM * DIM;
  _Float16* wvh = wkh + (size_t)DIM * KVDIM;
  float* q32 = (float*)(wvh + (size_t)DIM * KVDIM);
  float* k32 = q32 + (size_t)SEQ * DIM;
  float* v32 = k32 + (size_t)SEQ * KVDIM;
  _Float16* q16 = (_Float16*)(v32 + (size_t)SEQ * KVDIM);
  _Float16* k16 = q16 + (size_t)SEQ * DIM;
  _Float16* v16 = k16 + (size_t)SEQ * KVDIM;
  _Float16* attnh = v16 + (size_t)SEQ * KVDIM;

  cvt_f16_kernel<<<(SEQ * DIM) / 1024, 256, 0, stream>>>(x, xh);
  cvt_f16_kernel<<<(DIM * DIM) / 1024, 256, 0, stream>>>(wq, wqh);
  cvt_f16_kernel<<<(DIM * KVDIM) / 1024, 256, 0, stream>>>(wk, wkh);
  cvt_f16_kernel<<<(DIM * KVDIM) / 1024, 256, 0, stream>>>(wv, wvh);

  gemm_f16_wmma<DIM, DIM><<<dim3(DIM / 128, SEQ / 64), 256, 0, stream>>>(xh, wqh, q32);
  gemm_f16_wmma<KVDIM, DIM><<<dim3(KVDIM / 128, SEQ / 64), 256, 0, stream>>>(xh, wkh, k32);
  gemm_f16_wmma<KVDIM, DIM><<<dim3(KVDIM / 128, SEQ / 64), 256, 0, stream>>>(xh, wvh, v32);

  cvt_f16_kernel<<<(DIM * DIM) / 1024, 256, 0, stream>>>(wo, wqh);  // wqh now = wo f16

  rope_q_kernel<<<(SEQ * DIM / 2) / 256, 256, 0, stream>>>(q32, q16);
  rope_k_kernel<<<(SEQ * KVDIM / 2) / 256, 256, 0, stream>>>(k32, k16, kout);
  v_expand_kernel<<<(SEQ * KVDIM) / 256, 256, 0, stream>>>(v32, v16, vout);

  flash_kernel<<<dim3(SEQ / 128, NH), 256, 0, stream>>>(q16, k16, v16, attnh);

  gemm_f16_wmma<DIM, DIM><<<dim3(DIM / 128, SEQ / 64), 256, 0, stream>>>(attnh, wqh, out0);
}
